// RNN_60292750901480
// MI455X (gfx1250) — matrix operand order repaired and checked by reading, not compiled, not run on hardware
//
#include <hip/hip_runtime.h>
#include <stdint.h>

typedef __bf16 bf16;
typedef __attribute__((ext_vector_type(16))) __bf16 bf16x16;
typedef __attribute__((ext_vector_type(8)))  __bf16 bf16x8;
typedef __attribute__((ext_vector_type(8)))  float  f32x8;
typedef __attribute__((ext_vector_type(4)))  float  f32x4;

#define DIM_D 512
#define DIM_H 512
#define DIM_T 512
#define DIM_N 128
#define SH    520   // padded LDS row stride (halfs): row stride = 4 banks -> conflict-free A reads

__device__ __forceinline__ float tanh_fast(float x) {
#if __has_builtin(__builtin_amdgcn_tanhf)
    return __builtin_amdgcn_tanhf(x);
#elif __has_builtin(__builtin_amdgcn_tanh_f32)
    return __builtin_amdgcn_tanh_f32(x);
#else
    float xc = fminf(fmaxf(x, -15.f), 15.f);
    float e  = __expf(2.f * xc);
    return (e - 1.f) / (e + 1.f);
#endif
}

__device__ __forceinline__ uint32_t lds_off(const void* p) {
    // generic pointers to LDS keep the allocation-relative offset in addr[31:0]
    return (uint32_t)(uintptr_t)p;
}

// GLOBAL_LOAD_ASYNC_TO_LDS_B128 (GVS mode): mem = SADDR + VADDR + IOFFSET,
// LDS dest = LDS_BASE + VDST.  Tracked by ASYNCcnt.
__device__ __forceinline__ void async_b128_to_lds(uint32_t lds_byte_off,
                                                  const void* sbase,
                                                  uint32_t vbyte_off) {
    asm volatile("global_load_async_to_lds_b128 %0, %1, %2"
                 :: "v"(lds_byte_off), "v"(vbyte_off), "s"(sbase)
                 : "memory");
}
__device__ __forceinline__ void wait_async0() {
    asm volatile("s_wait_asynccnt 0x0" ::: "memory");
}

__device__ __forceinline__ bf16x16 make_a_frag(const bf16* p) {
    bf16x8 lo = *(const bf16x8*)p;
    bf16x8 hi = *(const bf16x8*)(p + 8);
    return __builtin_shufflevector(lo, hi, 0,1,2,3,4,5,6,7,8,9,10,11,12,13,14,15);
}

// ---------------------------------------------------------------------------
// Kernel 0: convert + transpose weights to bf16.  WxT[h][d] / WhT[h][k]
// ---------------------------------------------------------------------------
__global__ __launch_bounds__(256) void prep_weights(const float* __restrict__ Wx,
                                                    const float* __restrict__ Wh,
                                                    bf16* __restrict__ WxT,
                                                    bf16* __restrict__ WhT) {
    int idx = blockIdx.x * 256 + threadIdx.x;
    int sel = idx >= (DIM_D * DIM_H);
    int i   = sel ? idx - DIM_D * DIM_H : idx;
    int n   = i >> 9;
    int d   = i & 511;
    if (sel) WhT[i] = (bf16)Wh[d * DIM_H + n];
    else     WxT[i] = (bf16)Wx[d * DIM_H + n];
}

// ---------------------------------------------------------------------------
// Kernel 1: xW = x @ Wx + b  (M=65536, K=512, N=512) -> d_out
// x tile async-copied to LDS (f32), converted to bf16 LDS, K-loop pipelined.
// ---------------------------------------------------------------------------
__global__ __launch_bounds__(256) void xw_gemm(const float* __restrict__ x,
                                               const float* __restrict__ bias,
                                               const bf16* __restrict__ WxT,
                                               float* __restrict__ out) {
    __shared__ float xf[16 * DIM_D];   // raw f32 tile, 32 KB
    __shared__ bf16  xa[16 * SH];      // bf16 tile for WMMA A fragments
    const int  tid   = threadIdx.x;
    const long mbase = (long)blockIdx.x * 16;

    // ---- async stage: 16x512 f32 tile of x -> LDS (ASYNCcnt path) ----
    {
        const float* gp = x + mbase * DIM_D;      // uniform SGPR base
        uint32_t dst  = lds_off(xf);
        uint32_t toff = (uint32_t)tid * 128;      // 256 thr * 128 B = 32 KB
        #pragma unroll
        for (int i = 0; i < 8; ++i)
            async_b128_to_lds(dst + toff + i * 16, gp, toff + i * 16);
        wait_async0();
    }
    __syncthreads();

    // ---- convert f32 LDS -> bf16 LDS ----
    {
        int row = (tid * 32) >> 9;
        int col = (tid * 32) & 511;
        const float* src = xf + tid * 32;
        bf16* dst = xa + row * SH + col;
        #pragma unroll
        for (int c = 0; c < 4; ++c) {
            f32x4 f0 = *(const f32x4*)(src + c * 8);
            f32x4 f1 = *(const f32x4*)(src + c * 8 + 4);
            bf16x8 p;
            p[0] = (bf16)f0[0]; p[1] = (bf16)f0[1]; p[2] = (bf16)f0[2]; p[3] = (bf16)f0[3];
            p[4] = (bf16)f1[0]; p[5] = (bf16)f1[1]; p[6] = (bf16)f1[2]; p[7] = (bf16)f1[3];
            *(bf16x8*)(dst + c * 8) = p;
        }
    }
    __syncthreads();

    const int lane  = tid & 31;
    const int wave  = tid >> 5;
    const int m     = lane & 15;
    const int halfk = lane >> 4;
    const int cbase = wave * 64;

    f32x8 acc[4];
    #pragma unroll
    for (int c = 0; c < 4; ++c) {
        float bv = bias[cbase + c * 16 + m];
        #pragma unroll
        for (int r = 0; r < 8; ++r) acc[c][r] = bv;
    }

    const bf16* arow = xa + m * SH + halfk * 16;
    const bf16* bcol = WxT + (long)(cbase + m) * DIM_D + halfk * 16;

    // ---- software-pipelined K loop: fetch kt+1 while computing kt ----
    bf16x16 a = make_a_frag(arow);
    bf16x16 bcur[4];
    #pragma unroll
    for (int c = 0; c < 4; ++c)
        bcur[c] = *(const bf16x16*)(bcol + (long)c * 16 * DIM_D);

    #pragma unroll
    for (int kt = 0; kt < 16; ++kt) {
        bf16x16 an;
        bf16x16 bn[4];
        if (kt < 15) {
            an = make_a_frag(arow + (kt + 1) * 32);
            #pragma unroll
            for (int c = 0; c < 4; ++c)
                bn[c] = *(const bf16x16*)(bcol + (long)c * 16 * DIM_D + (kt + 1) * 32);
        }
        #pragma unroll
        for (int c = 0; c < 4; ++c)
            acc[c] = __builtin_amdgcn_wmma_f32_16x16x32_bf16(
                false, a, false, bcur[c], (short)0, acc[c], false, false);
        if (kt < 15) {
            a = an;
            #pragma unroll
            for (int c = 0; c < 4; ++c) bcur[c] = bn[c];
        }
    }

    #pragma unroll
    for (int c = 0; c < 4; ++c) {
        int col = cbase + c * 16 + m;
        #pragma unroll
        for (int r = 0; r < 8; ++r) {
            long row = mbase + r + 8 * halfk;
            out[row * DIM_H + col] = acc[c][r];
        }
    }
}

// ---------------------------------------------------------------------------
// Kernel 2: recurrence.  8 blocks (16 batch rows each), 1024 thr (32 waves,
// one 16x16 column tile each).  Wh fragments preloaded into registers once;
// h double-buffered in LDS bf16; xW reads hoisted + next-step prefetch.
// ---------------------------------------------------------------------------
__global__ __launch_bounds__(1024) void rnn_scan(const float* __restrict__ h0,
                                                 const bf16* __restrict__ WhT,
                                                 float* __restrict__ out) {
    __shared__ bf16 hbuf[2][16 * SH];
    const int tid = threadIdx.x;
    const int g   = blockIdx.x;

    for (int i = tid; i < 16 * DIM_H; i += 1024) {
        int row = i >> 9, col = i & 511;
        hbuf[0][row * SH + col] = (bf16)h0[(g * 16 + row) * DIM_H + col];
    }
    __syncthreads();

    const int lane  = tid & 31;
    const int wave  = tid >> 5;
    const int m     = lane & 15;
    const int halfk = lane >> 4;
    const int col   = wave * 16 + m;

    // ---- preload this wave's Wh slice: 16 B-fragments = 128 VGPRs ----
    const bf16* wrow = WhT + (long)col * DIM_H + halfk * 16;
    bf16x16 bfr[16];
    #pragma unroll
    for (int kt = 0; kt < 16; ++kt)
        bfr[kt] = *(const bf16x16*)(wrow + kt * 32);

    const long row0   = (long)(g * 16 + 8 * halfk);   // r=0 output row
    const long rowstr = (long)DIM_T * DIM_H;          // between consecutive m

    int buf = 0;
    for (int t = 0; t < DIM_T; ++t) {
        // issue xW loads early so they overlap the WMMA chain
        float xw[8];
        const float* op = out + (row0 * DIM_T + t) * DIM_H + col;
        #pragma unroll
        for (int r = 0; r < 8; ++r) xw[r] = op[(long)r * rowstr];

        f32x8 acc0 = {0.f,0.f,0.f,0.f,0.f,0.f,0.f,0.f};
        f32x8 acc1 = {0.f,0.f,0.f,0.f,0.f,0.f,0.f,0.f};
        const bf16* hrow = hbuf[buf] + m * SH + halfk * 16;
        #pragma unroll
        for (int kt = 0; kt < 16; kt += 2) {
            acc0 = __builtin_amdgcn_wmma_f32_16x16x32_bf16(
                false, make_a_frag(hrow + kt * 32), false, bfr[kt],
                (short)0, acc0, false, false);
            acc1 = __builtin_amdgcn_wmma_f32_16x16x32_bf16(
                false, make_a_frag(hrow + (kt + 1) * 32), false, bfr[kt + 1],
                (short)0, acc1, false, false);
        }

        bf16* hnext = hbuf[buf ^ 1];
        float* wp = (float*)op;
        #pragma unroll
        for (int r = 0; r < 8; ++r) {
            int   mrow = r + 8 * halfk;
            float v    = tanh_fast(acc0[r] + acc1[r] + xw[r]);
            wp[(long)r * rowstr] = v;
            hnext[mrow * SH + col] = (bf16)v;
            __builtin_prefetch(wp + (long)r * rowstr + DIM_H, 0, 0);  // next t's xW
        }
        __syncthreads();
        buf ^= 1;
    }
}

extern "C" void kernel_launch(void* const* d_in, const int* in_sizes, int n_in,
                              void* d_out, int out_size, void* d_ws, size_t ws_size,
                              hipStream_t stream) {
    const float* x  = (const float*)d_in[0];
    const float* h0 = (const float*)d_in[1];
    const float* Wx = (const float*)d_in[2];
    const float* Wh = (const float*)d_in[3];
    const float* b  = (const float*)d_in[4];
    float* out = (float*)d_out;

    bf16* WxT = (bf16*)d_ws;
    bf16* WhT = WxT + DIM_D * DIM_H;

    prep_weights<<<2 * DIM_D * DIM_H / 256, 256, 0, stream>>>(Wx, Wh, WxT, WhT);
    xw_gemm<<<(DIM_N * DIM_T) / 16, 256, 0, stream>>>(x, b, WxT, out);
    rnn_scan<<<DIM_N / 16, 1024, 0, stream>>>(h0, WhT, out);
}
